// Block_25683904430373
// MI455X (gfx1250) — matrix operand order repaired and checked by reading, not compiled, not run on hardware
//
#include <hip/hip_runtime.h>
#include <hip/hip_bf16.h>

// ---------------------------------------------------------------------------
// Types for CDNA5 WMMA
// ---------------------------------------------------------------------------
typedef __attribute__((ext_vector_type(16))) __bf16 v16bf;
typedef __attribute__((ext_vector_type(8)))  float  v8f;

union FragBF16 {
    uint4 q[2];
    v16bf v;
};

__device__ __forceinline__ unsigned short f2bf(float f) {
    union { float f; unsigned u; } v;
    v.f = f;
    unsigned r = v.u + 0x7FFFu + ((v.u >> 16) & 1u);   // round-to-nearest-even
    return (unsigned short)(r >> 16);
}

// CDNA5 async global->LDS copy (16B per lane), tracked by ASYNCcnt.
// LDS byte address = low 32 bits of the flat shared pointer (ISA 10.2: LDS
// aperture maps addr[31:0] as the LDS offset).
__device__ __forceinline__ void async_copy_b128(void* lds_dst, const void* gsrc) {
    unsigned loff = (unsigned)(size_t)lds_dst;
    asm volatile("global_load_async_to_lds_b128 %0, %1, off"
                 :: "v"(loff), "v"(gsrc) : "memory");
}
__device__ __forceinline__ void wait_async0() {
    asm volatile("s_wait_asynccnt 0" ::: "memory");
}

// ---------------------------------------------------------------------------
// LayerNorm (fp32 in) -> bf16 out.  One block (256 thr) per row of D=1024.
// ---------------------------------------------------------------------------
__global__ __launch_bounds__(256)
void layernorm_bf16_k(const float* __restrict__ x,
                      const float* __restrict__ g,
                      const float* __restrict__ b,
                      unsigned short* __restrict__ out, int D) {
    const float* row = x + (size_t)blockIdx.x * D;
    unsigned short* orow = out + (size_t)blockIdx.x * D;
    __shared__ float red[256];
    int tid = threadIdx.x;

    float s = 0.f;
    for (int i = tid; i < D; i += 256) s += row[i];
    red[tid] = s; __syncthreads();
    for (int k = 128; k > 0; k >>= 1) { if (tid < k) red[tid] += red[tid + k]; __syncthreads(); }
    float mu = red[0] / (float)D;
    __syncthreads();

    float v = 0.f;
    for (int i = tid; i < D; i += 256) { float d = row[i] - mu; v += d * d; }
    red[tid] = v; __syncthreads();
    for (int k = 128; k > 0; k >>= 1) { if (tid < k) red[tid] += red[tid + k]; __syncthreads(); }
    float rstd = rsqrtf(red[0] / (float)D + 1e-5f);

    for (int i = tid; i < D; i += 256)
        orow[i] = f2bf((row[i] - mu) * rstd * g[i] + b[i]);
}

// ---------------------------------------------------------------------------
// Row softmax over fp32 scores -> bf16 probabilities.  One block per row.
// ---------------------------------------------------------------------------
__global__ __launch_bounds__(256)
void softmax_rows_k(const float* __restrict__ S,
                    unsigned short* __restrict__ P, int N) {
    const float* row = S + (size_t)blockIdx.x * N;
    unsigned short* orow = P + (size_t)blockIdx.x * N;
    __shared__ float red[256];
    int tid = threadIdx.x;

    float m = -3.4e38f;
    for (int i = tid; i < N; i += 256) m = fmaxf(m, row[i]);
    red[tid] = m; __syncthreads();
    for (int k = 128; k > 0; k >>= 1) { if (tid < k) red[tid] = fmaxf(red[tid], red[tid + k]); __syncthreads(); }
    m = red[0]; __syncthreads();

    float s = 0.f;
    for (int i = tid; i < N; i += 256) s += __expf(row[i] - m);
    red[tid] = s; __syncthreads();
    for (int k = 128; k > 0; k >>= 1) { if (tid < k) red[tid] += red[tid + k]; __syncthreads(); }
    float inv = 1.f / red[0];

    for (int i = tid; i < N; i += 256)
        orow[i] = f2bf(__expf(row[i] - m) * inv);
}

// ---------------------------------------------------------------------------
// Templated bf16 WMMA GEMM, double-buffered LDS, async global->LDS staging.
//   C[M,N] = epilogue( A[M,K](bf16) @ B[K,N] )
// Block: 256 threads = 8 waves; block tile 128(M) x 128(N); K-step 32.
// Wave grid 4(M) x 2(N); wave tile 32 x 64 -> 2x4 accumulators of 16x16.
//
// BMODE: 0 = B fp32 row-major [K,N] (weights; reg-staged, convert->bf16)
//        1 = B bf16 row-major [K,N] (async global->LDS)
//        2 = B bf16 row-major [N,K] (reg-staged, transposed into LDS)
// EPI:   0 = store bf16 (no bias)
//        1 = +bias, store bf16
//        2 = +bias, ReLU, store bf16
//        3 = +bias, +residual(fp32), store fp32
//        4 = *scale, causal mask (col>row -> -1e9), store fp32
// ---------------------------------------------------------------------------
template <int BMODE, int EPI>
__global__ __launch_bounds__(256, 2)
void gemm_bf16_wmma(const unsigned short* __restrict__ A, long long sAb, int lda,
                    const void* __restrict__ Bv, long long sBb, int ldb,
                    const float* __restrict__ bias,
                    const float* __restrict__ residual,
                    void* __restrict__ Cv, long long sCb, int ldc,
                    int M, int N, int K, float scale) {
    const int tid  = threadIdx.x;
    const int lane = tid & 31;
    const int w    = tid >> 5;
    const int wm   = (w >> 1) * 32;      // wave M offset within block tile
    const int wn   = (w & 1) * 64;       // wave N offset within block tile
    const int bm   = blockIdx.y * 128;
    const int bn   = blockIdx.x * 128;
    const int bz   = blockIdx.z;

    A += (size_t)bz * sAb;
    const float*          Bf = (const float*)Bv          + (size_t)bz * sBb;
    const unsigned short* Bh = (const unsigned short*)Bv + (size_t)bz * sBb;
    float*          Cf = (float*)Cv          + (size_t)bz * sCb;
    unsigned short* Ch = (unsigned short*)Cv + (size_t)bz * sCb;

    // Double-buffered padded LDS tiles (strides stagger banks, 16B aligned)
    __shared__ __align__(16) unsigned short As[2][128 * 40];  // [m][k]
    __shared__ __align__(16) unsigned short Bs[2][32 * 136];  // [k][n]

    v8f acc[2][4] = {};

    const int half = lane >> 4;   // A frag: which K-half this lane holds
    const int mrow = lane & 15;   // A frag: row within 16x16 tile

    const int ar  = tid >> 1, acs = (tid & 1) * 16;  // A staging: row, col-seg
    const int br  = tid >> 3, bcs = (tid & 7) * 16;  // B staging (N layouts)
    const int tn  = tid >> 1, tks = (tid & 1) * 16;  // B staging (T layout)

    // ---- prologue: stage K-tile 0 into buffer 0 ----
    {
        const unsigned short* srcA = A + (size_t)(bm + ar) * lda + acs;
        async_copy_b128(&As[0][ar * 40 + acs],     srcA);
        async_copy_b128(&As[0][ar * 40 + acs + 8], srcA + 8);
        if (BMODE == 0) {
            const float4* s4 = (const float4*)(Bf + (size_t)br * ldb + bn + bcs);
            float4 f0 = s4[0], f1 = s4[1], f2 = s4[2], f3 = s4[3];
            unsigned short* d = &Bs[0][br * 136 + bcs];
            d[0]  = f2bf(f0.x); d[1]  = f2bf(f0.y); d[2]  = f2bf(f0.z); d[3]  = f2bf(f0.w);
            d[4]  = f2bf(f1.x); d[5]  = f2bf(f1.y); d[6]  = f2bf(f1.z); d[7]  = f2bf(f1.w);
            d[8]  = f2bf(f2.x); d[9]  = f2bf(f2.y); d[10] = f2bf(f2.z); d[11] = f2bf(f2.w);
            d[12] = f2bf(f3.x); d[13] = f2bf(f3.y); d[14] = f2bf(f3.z); d[15] = f2bf(f3.w);
        } else if (BMODE == 1) {
            const unsigned short* srcB = Bh + (size_t)br * ldb + bn + bcs;
            async_copy_b128(&Bs[0][br * 136 + bcs],     srcB);
            async_copy_b128(&Bs[0][br * 136 + bcs + 8], srcB + 8);
        } else {
            const uint4* s4 = (const uint4*)(Bh + (size_t)(bn + tn) * ldb + tks);
            union { uint4 q[2]; unsigned short u[16]; } t;
            t.q[0] = s4[0]; t.q[1] = s4[1];
#pragma unroll
            for (int e = 0; e < 16; ++e) Bs[0][(tks + e) * 136 + tn] = t.u[e];
        }
        wait_async0();
        __syncthreads();
    }

    int buf = 0;
    for (int k0 = 0; k0 < K; k0 += 32, buf ^= 1) {
        const int  nk = k0 + 32;
        const bool hn = nk < K;
        unsigned short* asn = As[buf ^ 1];
        unsigned short* bsn = Bs[buf ^ 1];

        float4 bf4[4];                                      // BMODE0 staging
        union { uint4 q[2]; unsigned short u[16]; } bt;     // BMODE2 staging

        // ---- kick off staging of the NEXT K-tile ----
        if (hn) {
            const unsigned short* srcA = A + (size_t)(bm + ar) * lda + nk + acs;
            async_copy_b128(&asn[ar * 40 + acs],     srcA);
            async_copy_b128(&asn[ar * 40 + acs + 8], srcA + 8);
            if (BMODE == 0) {
                const float4* s4 = (const float4*)(Bf + (size_t)(nk + br) * ldb + bn + bcs);
                bf4[0] = s4[0]; bf4[1] = s4[1]; bf4[2] = s4[2]; bf4[3] = s4[3];
            } else if (BMODE == 1) {
                const unsigned short* srcB = Bh + (size_t)(nk + br) * ldb + bn + bcs;
                async_copy_b128(&bsn[br * 136 + bcs],     srcB);
                async_copy_b128(&bsn[br * 136 + bcs + 8], srcB + 8);
            } else {
                const uint4* s4 = (const uint4*)(Bh + (size_t)(bn + tn) * ldb + nk + tks);
                bt.q[0] = s4[0]; bt.q[1] = s4[1];
            }
        }

        // ---- fragments + WMMA on the CURRENT buffer ----
        const unsigned short* as = As[buf];
        const unsigned short* bs = Bs[buf];
        FragBF16 fa[2], fb[4];
#pragma unroll
        for (int i = 0; i < 2; ++i) {
            const unsigned short* p = &as[(wm + i * 16 + mrow) * 40 + half * 8];
            fa[i].q[0] = *(const uint4*)p;         // K = half*8 + 0..7
            fa[i].q[1] = *(const uint4*)(p + 16);  // K = 16 + half*8 + 0..7
        }
#pragma unroll
        for (int j = 0; j < 4; ++j) {
            const unsigned short* p = &bs[(half * 8) * 136 + wn + j * 16 + mrow];  // N=mrow, K as A frag
            unsigned short* d = (unsigned short*)&fb[j];
#pragma unroll
            for (int e = 0; e < 8; ++e) { d[e] = p[e * 136]; d[8 + e] = p[(16 + e) * 136]; }
        }
#pragma unroll
        for (int i = 0; i < 2; ++i)
#pragma unroll
            for (int j = 0; j < 4; ++j)
                acc[i][j] = __builtin_amdgcn_wmma_f32_16x16x32_bf16(
                    false, fa[i].v, false, fb[j].v, (short)0, acc[i][j], false, false);

        // ---- finish register-staged B for the next buffer ----
        if (hn) {
            if (BMODE == 0) {
                unsigned short* d = &bsn[br * 136 + bcs];
                d[0]  = f2bf(bf4[0].x); d[1]  = f2bf(bf4[0].y); d[2]  = f2bf(bf4[0].z); d[3]  = f2bf(bf4[0].w);
                d[4]  = f2bf(bf4[1].x); d[5]  = f2bf(bf4[1].y); d[6]  = f2bf(bf4[1].z); d[7]  = f2bf(bf4[1].w);
                d[8]  = f2bf(bf4[2].x); d[9]  = f2bf(bf4[2].y); d[10] = f2bf(bf4[2].z); d[11] = f2bf(bf4[2].w);
                d[12] = f2bf(bf4[3].x); d[13] = f2bf(bf4[3].y); d[14] = f2bf(bf4[3].z); d[15] = f2bf(bf4[3].w);
            } else if (BMODE == 2) {
#pragma unroll
                for (int e = 0; e < 16; ++e) bsn[(tks + e) * 136 + tn] = bt.u[e];
            }
        }
        wait_async0();
        __syncthreads();
    }

    // ---- epilogue.  C/D layout: col = lane%16, rows = 8*(lane/16)+vgpr ----
#pragma unroll
    for (int i = 0; i < 2; ++i) {
#pragma unroll
        for (int j = 0; j < 4; ++j) {
            int r0 = bm + wm + i * 16 + ((lane >> 4) << 3);
            int gc = bn + wn + j * 16 + (lane & 15);
            float bval = (EPI == 1 || EPI == 2 || EPI == 3) ? bias[gc] : 0.f;
#pragma unroll
            for (int e = 0; e < 8; ++e) {
                int gr = r0 + e;
                size_t idx = (size_t)gr * ldc + gc;
                float v = acc[i][j][e];
                if (EPI == 0) {
                    Ch[idx] = f2bf(v);
                } else if (EPI == 1) {
                    Ch[idx] = f2bf(v + bval);
                } else if (EPI == 2) {
                    Ch[idx] = f2bf(fmaxf(v + bval, 0.f));
                } else if (EPI == 3) {
                    Cf[idx] = v + bval + residual[idx];
                } else { // EPI == 4 : scaled causal scores
                    Cf[idx] = (gc > gr) ? -1e9f : v * scale;
                }
            }
        }
    }
}

// ---------------------------------------------------------------------------
// Host-side orchestration
// ---------------------------------------------------------------------------
extern "C" void kernel_launch(void* const* d_in, const int* in_sizes, int n_in,
                              void* d_out, int out_size, void* d_ws, size_t ws_size,
                              hipStream_t stream) {
    (void)in_sizes; (void)n_in; (void)out_size; (void)ws_size;

    const int B = 4, T = 2048, D = 1024, F = 4096;
    const int BT = B * T;                 // 8192 rows

    const float* x      = (const float*)d_in[0];
    const float* ln1_g  = (const float*)d_in[1];
    const float* ln1_b  = (const float*)d_in[2];
    const float* qkv_w  = (const float*)d_in[3];
    const float* qkv_b  = (const float*)d_in[4];
    const float* proj_w = (const float*)d_in[5];
    const float* proj_b = (const float*)d_in[6];
    const float* ln2_g  = (const float*)d_in[7];
    const float* ln2_b  = (const float*)d_in[8];
    const float* ff1_w  = (const float*)d_in[9];
    const float* ff1_b  = (const float*)d_in[10];
    const float* ff2_w  = (const float*)d_in[11];
    const float* ff2_b  = (const float*)d_in[12];
    float* out = (float*)d_out;

    // Workspace layout (bytes); regions reused across phases.
    char* ws = (char*)d_ws;
    const size_t o_h   = 0;                                 // h -> y -> h2 (bf16)
    const size_t o_qkv = o_h   + (size_t)BT * D * 2;        // qkv bf16
    const size_t o_S   = o_qkv + (size_t)BT * 3 * D * 2;    // S fp32 -> a1 bf16
    const size_t o_P   = o_S   + (size_t)B * T * T * 4;     // P bf16
    const size_t o_x1  = o_P   + (size_t)B * T * T * 2;     // x1 fp32

    unsigned short* h   = (unsigned short*)(ws + o_h);
    unsigned short* qkv = (unsigned short*)(ws + o_qkv);
    float*          S   = (float*)(ws + o_S);
    unsigned short* P   = (unsigned short*)(ws + o_P);
    unsigned short* y   = (unsigned short*)(ws + o_h);   // reuse (h dead)
    float*          x1  = (float*)(ws + o_x1);
    unsigned short* h2  = (unsigned short*)(ws + o_h);   // reuse (y dead)
    unsigned short* a1  = (unsigned short*)(ws + o_S);   // reuse (S dead)

    dim3 thr(256);

    // 1) LN1: h = LN(x) (bf16)
    layernorm_bf16_k<<<BT, thr, 0, stream>>>(x, ln1_g, ln1_b, h, D);

    // 2) qkv = h @ qkv_w + qkv_b   [8192 x 3072] bf16
    gemm_bf16_wmma<0, 1><<<dim3(3 * D / 128, BT / 128, 1), thr, 0, stream>>>(
        h, 0, D, qkv_w, 0, 3 * D, qkv_b, nullptr, qkv, 0, 3 * D,
        BT, 3 * D, D, 1.f);

    // 3) S = (Q @ K^T) / 32, causal-masked   [4 x 2048 x 2048] fp32 (batched)
    gemm_bf16_wmma<2, 4><<<dim3(T / 128, T / 128, B), thr, 0, stream>>>(
        qkv /*Q*/, (long long)T * 3 * D, 3 * D,
        qkv + D /*K*/, (long long)T * 3 * D, 3 * D,
        nullptr, nullptr, S, (long long)T * T, T,
        T, T, D, 0.03125f /* 1/sqrt(1024) */);

    // 4) P = softmax(S) (bf16)
    softmax_rows_k<<<B * T, thr, 0, stream>>>(S, P, T);

    // 5) y = P @ V   [4 x 2048 x 1024] bf16 (batched)
    gemm_bf16_wmma<1, 0><<<dim3(D / 128, T / 128, B), thr, 0, stream>>>(
        P, (long long)T * T, T,
        qkv + 2 * D /*V*/, (long long)T * 3 * D, 3 * D,
        nullptr, nullptr, y, (long long)T * D, D,
        T, D, T, 1.f);

    // 6) x1 = x + y @ proj_w + proj_b   fp32
    gemm_bf16_wmma<0, 3><<<dim3(D / 128, BT / 128, 1), thr, 0, stream>>>(
        y, 0, D, proj_w, 0, D, proj_b, x, x1, 0, D,
        BT, D, D, 1.f);

    // 7) LN2: h2 = LN(x1) (bf16)
    layernorm_bf16_k<<<BT, thr, 0, stream>>>(x1, ln2_g, ln2_b, h2, D);

    // 8) a1 = relu(h2 @ ff1_w + ff1_b)   [8192 x 4096] bf16
    gemm_bf16_wmma<0, 2><<<dim3(F / 128, BT / 128, 1), thr, 0, stream>>>(
        h2, 0, D, ff1_w, 0, F, ff1_b, nullptr, a1, 0, F,
        BT, F, D, 1.f);

    // 9) out = x1 + a1 @ ff2_w + ff2_b   fp32
    gemm_bf16_wmma<0, 3><<<dim3(D / 128, BT / 128, 1), thr, 0, stream>>>(
        a1, 0, F, ff2_w, 0, D, ff2_b, x1, out, 0, D,
        BT, D, F, 1.f);
}
